// GRUMCModel_8796093022918
// MI455X (gfx1250) — matrix operand order repaired and checked by reading, not compiled, not run on hardware
//
#include <hip/hip_runtime.h>
#include <math.h>

// ---------------------------------------------------------------------------
// MI455X (gfx1250) 13-layer residual bidirectional-GRU classifier.
// Roofline: xs read = 163.6MB (~7us @ 23.3TB/s) dominates the parallel part;
// each A-tile of xs is loaded exactly ONCE. Weights are pre-packed to f16 in
// WMMA fragment order so the hot loops load B operands as 2x b128 per frag
// with no per-iteration cvt. Serial part: 13*128 GRU steps, 12 WMMAs/step
// against register-resident Wr fragments, barrier-free LDS h-transpose.
// ---------------------------------------------------------------------------

typedef __attribute__((ext_vector_type(16))) _Float16 v16h;
typedef __attribute__((ext_vector_type(8)))  float    v8f;

__device__ __forceinline__ v8f wmma_f16(v16h a, v16h b, v8f c) {
  return __builtin_amdgcn_wmma_f32_16x16x32_f16(false, a, false, b, (short)0, c, false, false);
}

// ----- WMMA fragment builders (layouts per CDNA5 ISA 7.12.2) -----
// A (16x32 f16): lanes 0-15 -> M=lane, K = {0..7,16..23}; lanes 16-31 -> K+8
__device__ __forceinline__ v16h frag_a_from_f32(const float* p, int lda, int l, int hi) {
  v16h a;
#pragma unroll
  for (int hh = 0; hh < 16; ++hh) {
    int k = ((hh >> 3) << 4) | (hi << 3) | (hh & 7);
    a[hh] = (_Float16)p[l * lda + k];
  }
  return a;
}
__device__ __forceinline__ v16h frag_a_from_f16(const _Float16* p, int lda, int l, int hi) {
  v16h a;
#pragma unroll
  for (int hh = 0; hh < 16; ++hh) {
    int k = ((hh >> 3) << 4) | (hi << 3) | (hh & 7);
    a[hh] = p[l * lda + k];
  }
  return a;
}
// B (32x16 f16): lanes 0-15 -> N=lane, K = {0..7,16..23}; lanes 16-31 -> K+8
__device__ __forceinline__ v16h frag_b_from_f32(const float* p, int ldb, int l, int hi) {
  v16h b;
#pragma unroll
  for (int hh = 0; hh < 16; ++hh) {
    int k = ((hh >> 3) << 4) | (hi << 3) | (hh & 7);
    b[hh] = (_Float16)p[k * ldb + l];
  }
  return b;
}

// ---------------------------------------------------------------------------
// K0: zero seq [B*S*64] and h state [2*32*32]
__global__ void init_kernel(float* __restrict__ seq, float* __restrict__ hbuf) {
  int i = blockIdx.x * blockDim.x + threadIdx.x;
  if (i < 4096 * 64) seq[i] = 0.0f;
  if (i < 2048)      hbuf[i] = 0.0f;
}

// ---------------------------------------------------------------------------
// K0b: pre-pack W_lin [768,64] and Wk_{f,b} [64,96] to f16 in fragment order
// [kb][nt][lane][hh] -> B frag load becomes one contiguous 32B (2x b128).
__global__ void pack_kernel(const float* __restrict__ Wlin,
                            const float* __restrict__ Wk_f, const float* __restrict__ Wk_b,
                            _Float16* __restrict__ wlin_pk, _Float16* __restrict__ wk_pk) {
  int id = blockIdx.x * blockDim.x + threadIdx.x;
  if (id < 24 * 4 * 32 * 16) {                     // W_lin: 24 kb x 4 nt
    int hh = id & 15, lane = (id >> 4) & 31, nt = (id >> 9) & 3, kb = id >> 11;
    int l = lane & 15, hi = lane >> 4;
    int k = ((hh >> 3) << 4) | (hi << 3) | (hh & 7);
    wlin_pk[id] = (_Float16)Wlin[(size_t)(kb * 32 + k) * 64 + nt * 16 + l];
  }
  if (id < 2 * 2 * 6 * 32 * 16) {                  // Wk: dir x 2 kb x 6 nt
    int hh = id & 15, lane = (id >> 4) & 31;
    int rest = id >> 9;                            // dir*12 + kb*6 + nt
    int nt = rest % 6, kb = (rest / 6) & 1, dir = rest / 12;
    int l = lane & 15, hi = lane >> 4;
    int k = ((hh >> 3) << 4) | (hi << 3) | (hh & 7);
    const float* Wk = dir ? Wk_b : Wk_f;
    wk_pk[id] = (_Float16)Wk[(size_t)(kb * 32 + k) * 96 + nt * 16 + l];
  }
}

// ---------------------------------------------------------------------------
// K1: lin = xs @ W_lin + b_lin   [nrows,768]@[768,64] -> [nrows,64] f32
// One wave per 16-row band: A loaded ONCE, reused across 4 n-tile
// accumulators; B frags come pre-packed f16 (b128 loads, no cvt).
__global__ void __launch_bounds__(256)
lin_gemm_kernel(const float* __restrict__ xs, const v16h* __restrict__ wpk,
                const float* __restrict__ blin, float* __restrict__ lin, int nrows) {
  int wave = (blockIdx.x * blockDim.x + threadIdx.x) >> 5;
  int lane = threadIdx.x & 31;
  int l = lane & 15, hi = lane >> 4;
  if (wave * 16 >= nrows) return;                  // wave-uniform: EXEC stays full
  const float* abase = xs + (size_t)wave * 16 * 768;
  v8f c[4] = {};
  for (int kb = 0; kb < 24; ++kb) {
    int k0 = kb * 32;
    __builtin_prefetch(abase + l * 768 + k0 + 96, 0, 3);   // global_prefetch_b8
    v16h a = frag_a_from_f32(abase + k0, 768, l, hi);
#pragma unroll
    for (int nt = 0; nt < 4; ++nt) {
      v16h b = wpk[(size_t)(kb * 4 + nt) * 32 + lane];
      c[nt] = wmma_f16(a, b, c[nt]);
    }
  }
#pragma unroll
  for (int nt = 0; nt < 4; ++nt) {
    float bias = blin[nt * 16 + l];
#pragma unroll
    for (int v = 0; v < 8; ++v) {
      size_t row = (size_t)wave * 16 + v + hi * 8;
      lin[row * 64 + nt * 16 + l] = c[nt][v] + bias;
    }
  }
}

// ---------------------------------------------------------------------------
// K2: x16 = f16( LayerNorm(lin + seq) )   wave-per-row over G=64
__global__ void __launch_bounds__(256)
ln_kernel(const float* __restrict__ lin, const float* __restrict__ seq,
          const float* __restrict__ gamma, const float* __restrict__ beta,
          _Float16* __restrict__ x16) {
  int row  = (blockIdx.x * blockDim.x + threadIdx.x) >> 5;
  int lane = threadIdx.x & 31;
  if (row >= 4096) return;
  int j0 = lane * 2;
  float v0 = lin[(size_t)row * 64 + j0]     + seq[(size_t)row * 64 + j0];
  float v1 = lin[(size_t)row * 64 + j0 + 1] + seq[(size_t)row * 64 + j0 + 1];
  float s = v0 + v1, ss = v0 * v0 + v1 * v1;
#pragma unroll
  for (int m = 16; m >= 1; m >>= 1) {
    s  += __shfl_xor(s,  m, 32);
    ss += __shfl_xor(ss, m, 32);
  }
  float mean = s * (1.0f / 64.0f);
  float var  = ss * (1.0f / 64.0f) - mean * mean;
  float rs   = rsqrtf(var + 1e-3f);
  x16[(size_t)row * 64 + j0]     = (_Float16)((v0 - mean) * rs * gamma[j0]     + beta[j0]);
  x16[(size_t)row * 64 + j0 + 1] = (_Float16)((v1 - mean) * rs * gamma[j0 + 1] + beta[j0 + 1]);
}

// ---------------------------------------------------------------------------
// K3: xp_{f,b} = x @ Wk_{f,b} + b[0]   [4096,64]@[64,96] for both directions.
// One wave per 16-row band, 12 accumulators; B frags pre-packed f16.
// Output stored TRANSPOSED as [S,96,B] so the scan reads b128-contiguous.
__global__ void __launch_bounds__(256)
xp_gemm_kernel(const _Float16* __restrict__ x16, const v16h* __restrict__ wkpk,
               const float* __restrict__ bf, const float* __restrict__ bb,
               float* __restrict__ xp_f, float* __restrict__ xp_b) {
  int wave = (blockIdx.x * blockDim.x + threadIdx.x) >> 5;
  int lane = threadIdx.x & 31;
  int l = lane & 15, hi = lane >> 4;
  if (wave >= 256) return;
  v8f c[12] = {};
#pragma unroll
  for (int kb = 0; kb < 2; ++kb) {
    v16h a = frag_a_from_f16(x16 + (size_t)(wave * 16) * 64 + kb * 32, 64, l, hi);
#pragma unroll
    for (int nn = 0; nn < 12; ++nn) {
      int dir = nn / 6, nt = nn % 6;
      v16h b = wkpk[(size_t)(dir * 12 + kb * 6 + nt) * 32 + lane];
      c[nn] = wmma_f16(a, b, c[nn]);
    }
  }
#pragma unroll
  for (int nn = 0; nn < 12; ++nn) {
    int   dir  = nn / 6, nt = nn % 6;
    int   col  = nt * 16 + l;
    float bias = (dir ? bb : bf)[col];               // row 0 = input bias
    float* out = dir ? xp_b : xp_f;
#pragma unroll
    for (int v = 0; v < 8; ++v) {
      int gr   = wave * 16 + v + hi * 8;  // flat (b,s); 16-bands never cross batch
      int bidx = gr >> 7;
      int sidx = gr & 127;
      out[((size_t)sidx * 96 + col) * 32 + bidx] = c[nn][v] + bias;
    }
  }
}

// ---------------------------------------------------------------------------
// K4: serial GRU scan. 1 block, 2 waves: wave0=forward, wave1=backward.
// h in registers (D-frag layout); Wr as 6 resident B-frags. Per step:
// barrier-free LDS transpose of h (per-wave DS ordering suffices: each wave
// only touches its own hstage[dir] slice), 12 WMMAs, b128 xp loads, gates.
__global__ void __launch_bounds__(64)
scan_kernel(const float* __restrict__ xp_f, const float* __restrict__ xp_b,
            const float* __restrict__ Wr_f, const float* __restrict__ bf,
            const float* __restrict__ Wr_b, const float* __restrict__ bb,
            float* __restrict__ seq, float* __restrict__ hbuf) {
  __shared__ float hstage[2][32][33];
  int dir  = threadIdx.x >> 5;
  int lane = threadIdx.x & 31;
  int l = lane & 15, hi = lane >> 4;
  const float* xp    = dir ? xp_b : xp_f;
  const float* Wr    = dir ? Wr_b : Wr_f;
  const float* bias2 = dir ? bb   : bf;    // row 1 = recurrent bias
  float*       hst   = hbuf + dir * 1024;  // [32,32] f32 per direction

  v16h wb[6];
#pragma unroll
  for (int nt = 0; nt < 6; ++nt) wb[nt] = frag_b_from_f32(Wr + nt * 16, 96, l, hi);
  float rb[6];
#pragma unroll
  for (int nt = 0; nt < 6; ++nt) rb[nt] = bias2[96 + nt * 16 + l];

  v8f hfr[2][2];                            // D-frag layout: row=batch, col=unit
#pragma unroll
  for (int mt = 0; mt < 2; ++mt)
#pragma unroll
    for (int un = 0; un < 2; ++un)
#pragma unroll
      for (int v = 0; v < 8; ++v)
        hfr[mt][un][v] = hst[(mt * 16 + v + hi * 8) * 32 + un * 16 + l];

  for (int t = 0; t < 128; ++t) {
    int pos = dir ? (127 - t) : t;
    // stage h for transposed A-frag reload (own slice only -> no barrier)
#pragma unroll
    for (int mt = 0; mt < 2; ++mt)
#pragma unroll
      for (int un = 0; un < 2; ++un)
#pragma unroll
        for (int v = 0; v < 8; ++v)
          hstage[dir][mt * 16 + v + hi * 8][un * 16 + l] = hfr[mt][un][v];
    v16h am[2];
#pragma unroll
    for (int mt = 0; mt < 2; ++mt)
#pragma unroll
      for (int hh = 0; hh < 16; ++hh) {
        int k = ((hh >> 3) << 4) | (hi << 3) | (hh & 7);
        am[mt][hh] = (_Float16)hstage[dir][mt * 16 + l][k];
      }

#pragma unroll
    for (int mt = 0; mt < 2; ++mt) {
#pragma unroll
      for (int un = 0; un < 2; ++un) {
        v8f cz, cr, ch;
#pragma unroll
        for (int v = 0; v < 8; ++v) { cz[v] = rb[un]; cr[v] = rb[un + 2]; ch[v] = rb[un + 4]; }
        cz = wmma_f16(am[mt], wb[un],     cz);
        cr = wmma_f16(am[mt], wb[un + 2], cr);
        ch = wmma_f16(am[mt], wb[un + 4], ch);
        // xp tiles: b128 loads (32B-aligned thanks to [S,96,B] layout)
        const float4* xz4 = (const float4*)(xp + ((size_t)pos * 96 + un * 16      + l) * 32 + mt * 16 + hi * 8);
        const float4* xr4 = (const float4*)(xp + ((size_t)pos * 96 + un * 16 + 32 + l) * 32 + mt * 16 + hi * 8);
        const float4* xh4 = (const float4*)(xp + ((size_t)pos * 96 + un * 16 + 64 + l) * 32 + mt * 16 + hi * 8);
        float4 za = xz4[0], zb = xz4[1];
        float4 ra = xr4[0], rc = xr4[1];
        float4 ha = xh4[0], hb = xh4[1];
        float xzv[8] = {za.x, za.y, za.z, za.w, zb.x, zb.y, zb.z, zb.w};
        float xrv[8] = {ra.x, ra.y, ra.z, ra.w, rc.x, rc.y, rc.z, rc.w};
        float xhv[8] = {ha.x, ha.y, ha.z, ha.w, hb.x, hb.y, hb.z, hb.w};
        v8f hn;
#pragma unroll
        for (int v = 0; v < 8; ++v) {
          float z  = 1.0f / (1.0f + __expf(-(xzv[v] + cz[v])));
          float r  = 1.0f / (1.0f + __expf(-(xrv[v] + cr[v])));
          float hc = tanhf(xhv[v] + r * ch[v]);
          hn[v] = z * hfr[mt][un][v] + (1.0f - z) * hc;
        }
        hfr[mt][un] = hn;
#pragma unroll
        for (int v = 0; v < 8; ++v) {
          int batch = mt * 16 + v + hi * 8;
          seq[((size_t)batch * 128 + pos) * 64 + dir * 32 + un * 16 + l] = hn[v];
        }
      }
    }
  }
#pragma unroll
  for (int mt = 0; mt < 2; ++mt)
#pragma unroll
    for (int un = 0; un < 2; ++un)
#pragma unroll
      for (int v = 0; v < 8; ++v)
        hst[(mt * 16 + v + hi * 8) * 32 + un * 16 + l] = hfr[mt][un][v];
}

// ---------------------------------------------------------------------------
// K5: interleave(h1,h2) @ W_cls + b_cls -> softmax. lane = batch.
__global__ void cls_kernel(const float* __restrict__ hbuf, const float* __restrict__ Wcls,
                           const float* __restrict__ bcls, float* __restrict__ out) {
  int b = threadIdx.x;
  if (b >= 32) return;
  float a0 = bcls[0], a1 = bcls[1];
#pragma unroll
  for (int u = 0; u < 32; ++u) {
    float h1 = hbuf[b * 32 + u];
    float h2 = hbuf[1024 + b * 32 + u];
    a0 += h1 * Wcls[(2 * u) * 2 + 0] + h2 * Wcls[(2 * u + 1) * 2 + 0];
    a1 += h1 * Wcls[(2 * u) * 2 + 1] + h2 * Wcls[(2 * u + 1) * 2 + 1];
  }
  float mx = fmaxf(a0, a1);
  float e0 = __expf(a0 - mx), e1 = __expf(a1 - mx);
  float inv = 1.0f / (e0 + e1);
  out[b * 2 + 0] = e0 * inv;
  out[b * 2 + 1] = e1 * inv;
}

// ---------------------------------------------------------------------------
extern "C" void kernel_launch(void* const* d_in, const int* in_sizes, int n_in,
                              void* d_out, int out_size, void* d_ws, size_t ws_size,
                              hipStream_t stream) {
  (void)in_sizes; (void)n_in; (void)out_size;
  const float* xs    = (const float*)d_in[0];
  const float* Wlin  = (const float*)d_in[1];
  const float* blin  = (const float*)d_in[2];
  const float* gamma = (const float*)d_in[3];
  const float* beta  = (const float*)d_in[4];
  const float* Wk_f  = (const float*)d_in[5];
  const float* Wr_f  = (const float*)d_in[6];
  const float* b_f   = (const float*)d_in[7];
  const float* Wk_b  = (const float*)d_in[8];
  const float* Wr_b  = (const float*)d_in[9];
  const float* b_b   = (const float*)d_in[10];
  const float* Wcls  = (const float*)d_in[11];
  const float* bcls  = (const float*)d_in[12];
  float* out = (float*)d_out;

  // Workspace: if it fits, compute all 13 layers' lin in ONE streaming GEMM
  // launch (3328 waves saturating HBM); otherwise per-layer (deterministic
  // host-side choice: ws_size is fixed across calls).
  const size_t WLIN_PK = (size_t)24 * 4 * 32 * 16;     // f16 elements
  const size_t WK_PK   = (size_t)2 * 2 * 6 * 32 * 16;  // f16 elements
  const size_t LIN_ROWS_ALL = (size_t)13 * 4096;
  const size_t restBytes = (size_t)4096 * 64 * 2          // x16
                         + (size_t)2 * 128 * 96 * 32 * 4  // xp_f, xp_b
                         + (size_t)4096 * 64 * 4          // seq
                         + (size_t)2048 * 4               // hbuf
                         + (WLIN_PK + WK_PK) * 2;         // packed weights
  bool big = ws_size >= (LIN_ROWS_ALL * 64 * 4 + restBytes);
  size_t linBytes = (big ? LIN_ROWS_ALL : (size_t)4096) * 64 * 4;

  char*     ws      = (char*)d_ws;
  float*    lin     = (float*)ws;
  _Float16* x16     = (_Float16*)(ws + linBytes);
  float*    xp_f    = (float*)(ws + linBytes + (size_t)4096 * 64 * 2);
  float*    xp_b    = xp_f + 128 * 96 * 32;
  float*    seq     = xp_b + 128 * 96 * 32;
  float*    hbuf    = seq + 4096 * 64;
  _Float16* wlin_pk = (_Float16*)(hbuf + 2048);
  _Float16* wk_pk   = wlin_pk + WLIN_PK;

  init_kernel<<<1024, 256, 0, stream>>>(seq, hbuf);
  pack_kernel<<<192, 256, 0, stream>>>(Wlin, Wk_f, Wk_b, wlin_pk, wk_pk);
  if (big) {  // all layers at once: 53248/16 = 3328 waves -> 416 blocks
    lin_gemm_kernel<<<416, 256, 0, stream>>>(xs, (const v16h*)wlin_pk, blin, lin, 13 * 4096);
  }
  for (int i = 0; i < 13; ++i) {
    const float* lin_i = big ? (lin + (size_t)i * 4096 * 64) : lin;
    if (!big) {
      const float* xs_i = xs + (size_t)i * 32 * 128 * 768;
      lin_gemm_kernel<<<32, 256, 0, stream>>>(xs_i, (const v16h*)wlin_pk, blin, lin, 4096);
    }
    ln_kernel<<<512, 256, 0, stream>>>(lin_i, seq, gamma, beta, x16);
    xp_gemm_kernel<<<32, 256, 0, stream>>>(x16, (const v16h*)wk_pk, b_f, b_b, xp_f, xp_b);
    scan_kernel<<<1, 64, 0, stream>>>(xp_f, xp_b, Wr_f, b_f, Wr_b, b_b, seq, hbuf);
  }
  cls_kernel<<<1, 32, 0, stream>>>(hbuf, Wcls, bcls, out);
}
